// TemporalAttention_78847009620259
// MI455X (gfx1250) — matrix operand order repaired and checked by reading, not compiled, not run on hardware
//
#include <hip/hip_runtime.h>
#include <hip/hip_bf16.h>

// ---------------- problem constants ----------------
#define BB   64
#define TT   1024
#define DD   512
#define MTOT (BB * TT)          // 65536 rows for QKV projection
#define NEG_INF (-3.402823466e38f)

typedef __attribute__((ext_vector_type(16))) _Float16 v16h;
typedef __attribute__((ext_vector_type(8)))  _Float16 v8h;
typedef __attribute__((ext_vector_type(4)))  _Float16 v4h;
typedef __attribute__((ext_vector_type(8)))  float    v8f;

union V16U { v16h v; v8h h[2]; };

// A-fragment (16x32 f16): lane = row (lane&15); halves 0..7 at k+hs*8, 8..15 at k+16+hs*8
static __device__ __forceinline__ v16h afrag(const _Float16* rowk, int hs) {
  V16U u;
  u.h[0] = *(const v8h*)(rowk + hs * 8);
  u.h[1] = *(const v8h*)(rowk + 16 + hs * 8);
  return u.v;
}
// B-fragment (32x16 f16): lane = col (lane&15); halves 0..7 at k+hs*8, 8..15 at k+16+hs*8
static __device__ __forceinline__ v16h bfrag(const _Float16* colk, int hs) {
  V16U u;
  const _Float16* p = colk + hs * 8;
  u.h[0] = *(const v8h*)(p);
  u.h[1] = *(const v8h*)(p + 16);
  return u.v;
}

#define WMMA_F16(A, Bm, C) \
  __builtin_amdgcn_wmma_f32_16x16x32_f16(false, (A), false, (Bm), (short)0, (C), false, false)

// ---------------- setup kernels ----------------
__global__ void k_zero_out(float* __restrict__ out, int n) {
  int i = blockIdx.x * blockDim.x + threadIdx.x;
  if (i < n) out[i] = 0.0f;
}

__global__ void k_cvt_x(const float* __restrict__ x, _Float16* __restrict__ xh, int n) {
  int i = (blockIdx.x * blockDim.x + threadIdx.x) * 4;
  if (i + 3 < n) {
    float4 f = *(const float4*)(x + i);
    v4h o = { (_Float16)f.x, (_Float16)f.y, (_Float16)f.z, (_Float16)f.w };
    *(v4h*)(xh + i) = o;
  }
}

// W[k][n] (row-major) -> Wt[z][n][k] f16 (transposed, so B-fragments are contiguous in k)
__global__ void k_cvt_w(const float* __restrict__ Wq, const float* __restrict__ Wk,
                        const float* __restrict__ Wv, _Float16* __restrict__ Wt) {
  int idx = blockIdx.x * blockDim.x + threadIdx.x;
  if (idx >= 3 * DD * DD) return;
  int z = idx / (DD * DD);
  int rem = idx - z * DD * DD;
  int n = rem / DD, k = rem - n * DD;
  const float* W = (z == 0) ? Wq : ((z == 1) ? Wk : Wv);
  Wt[(size_t)z * DD * DD + (size_t)n * DD + k] = (_Float16)W[(size_t)k * DD + n];
}

// ---------------- QKV projection GEMM (WMMA f16 -> f32 -> f16) ----------------
// grid: (MTOT/256, DD/64, 3); block: 256 (8 waves). Wave computes a 32x64 tile
// (2 M-tiles x 4 N-tiles): B-fragments reused across both M-tiles.
// z==0 -> Qh [m][d], z==1 -> Kh [m][d], z==2 -> Vt [b][d][t] (transposed store).
__global__ __launch_bounds__(256) void k_qkv(
    const _Float16* __restrict__ xh, const _Float16* __restrict__ Wt,
    const float* __restrict__ bq, const float* __restrict__ bk, const float* __restrict__ bv,
    _Float16* __restrict__ Qh, _Float16* __restrict__ Kh, _Float16* __restrict__ Vt) {
  const int lane = threadIdx.x & 31;
  const int wave = threadIdx.x >> 5;
  const int hs = (lane < 16) ? 0 : 1;
  const int ln = lane & 15;
  const int z  = blockIdx.z;
  const int m0 = (blockIdx.x * 16 + wave * 2) * 16;   // first of two 16-row tiles
  const int nb = blockIdx.y * 64;

  const _Float16* Wz   = Wt + (size_t)z * DD * DD;
  const float*    bias = (z == 0) ? bq : ((z == 1) ? bk : bv);

  v8f c00 = {}, c01 = {}, c02 = {}, c03 = {};   // m-tile 0
  v8f c10 = {}, c11 = {}, c12 = {}, c13 = {};   // m-tile 1
  const _Float16* arow0 = xh + (size_t)(m0 + ln) * DD;
  const _Float16* arow1 = xh + (size_t)(m0 + 16 + ln) * DD;
  const _Float16* w0 = Wz + (size_t)(nb +  0 + ln) * DD;
  const _Float16* w1 = Wz + (size_t)(nb + 16 + ln) * DD;
  const _Float16* w2 = Wz + (size_t)(nb + 32 + ln) * DD;
  const _Float16* w3 = Wz + (size_t)(nb + 48 + ln) * DD;

  for (int kb = 0; kb < DD; kb += 32) {
    v16h aA = afrag(arow0 + kb, hs);
    v16h aB = afrag(arow1 + kb, hs);
    v16h b0 = bfrag(w0 + kb, hs);
    v16h b1 = bfrag(w1 + kb, hs);
    v16h b2 = bfrag(w2 + kb, hs);
    v16h b3 = bfrag(w3 + kb, hs);
    c00 = WMMA_F16(aA, b0, c00);
    c01 = WMMA_F16(aA, b1, c01);
    c02 = WMMA_F16(aA, b2, c02);
    c03 = WMMA_F16(aA, b3, c03);
    c10 = WMMA_F16(aB, b0, c10);
    c11 = WMMA_F16(aB, b1, c11);
    c12 = WMMA_F16(aB, b2, c12);
    c13 = WMMA_F16(aB, b3, c13);
  }

  const float bz0 = bias[nb +  0 + ln];
  const float bz1 = bias[nb + 16 + ln];
  const float bz2 = bias[nb + 32 + ln];
  const float bz3 = bias[nb + 48 + ln];

  for (int mt = 0; mt < 2; ++mt) {
    const v8f* s0 = mt ? &c10 : &c00;
    const v8f* s1 = mt ? &c11 : &c01;
    const v8f* s2 = mt ? &c12 : &c02;
    const v8f* s3 = mt ? &c13 : &c03;
    for (int r = 0; r < 8; ++r) {
      const int m = m0 + mt * 16 + r + hs * 8;
      if (z < 2) {
        _Float16* O = (z == 0) ? Qh : Kh;
        O[(size_t)m * DD + nb +  0 + ln] = (_Float16)((*s0)[r] + bz0);
        O[(size_t)m * DD + nb + 16 + ln] = (_Float16)((*s1)[r] + bz1);
        O[(size_t)m * DD + nb + 32 + ln] = (_Float16)((*s2)[r] + bz2);
        O[(size_t)m * DD + nb + 48 + ln] = (_Float16)((*s3)[r] + bz3);
      } else {
        const int b = m >> 10, t = m & (TT - 1);
        Vt[((size_t)b * DD + nb +  0 + ln) * TT + t] = (_Float16)((*s0)[r] + bz0);
        Vt[((size_t)b * DD + nb + 16 + ln) * TT + t] = (_Float16)((*s1)[r] + bz1);
        Vt[((size_t)b * DD + nb + 32 + ln) * TT + t] = (_Float16)((*s2)[r] + bz2);
        Vt[((size_t)b * DD + nb + 48 + ln) * TT + t] = (_Float16)((*s3)[r] + bz3);
      }
    }
  }
}

// ---------------- fused attention + mean ----------------
// grid: B * (T/16) workgroups; block 256 (8 waves). One (batch, 16-query tile) per WG.
__global__ __launch_bounds__(256) void k_attn(
    const _Float16* __restrict__ Qh, const _Float16* __restrict__ Kh,
    const _Float16* __restrict__ Vt, const unsigned char* __restrict__ mask,
    const float* __restrict__ scale_p, float* __restrict__ out) {
  __shared__ __align__(16) float    S[16 * TT];    // raw scores (64 KB)
  __shared__ __align__(16) _Float16 Ph[16 * TT];   // exp(s - max) f16 (32 KB)
  __shared__ float red[16 * 16];
  __shared__ float rowsum[16];

  const int tid  = threadIdx.x;
  const int lane = tid & 31;
  const int wave = tid >> 5;
  const int hs   = (lane < 16) ? 0 : 1;
  const int ln   = lane & 15;
  const int b    = blockIdx.x >> 6;
  const int qt   = blockIdx.x & 63;
  const int q0   = b * TT + qt * 16;      // global row of first query
  const float scale = scale_p[0];

  // ---- Phase 1: S[16 x 1024] = scale * Q_tile . K^T ----
  // Each wave: 8 key tiles, processed as 2 groups of 4 so one Q A-fragment
  // feeds 4 WMMA per d-step.
  const _Float16* qrow = Qh + (size_t)(q0 + ln) * DD;
  for (int g = 0; g < 2; ++g) {
    const int ktb = wave * 8 + g * 4;
    v8f s0 = {}, s1 = {}, s2 = {}, s3 = {};
    const _Float16* k0 = Kh + (size_t)(b * TT + (ktb + 0) * 16 + ln) * DD;
    const _Float16* k1 = Kh + (size_t)(b * TT + (ktb + 1) * 16 + ln) * DD;
    const _Float16* k2 = Kh + (size_t)(b * TT + (ktb + 2) * 16 + ln) * DD;
    const _Float16* k3 = Kh + (size_t)(b * TT + (ktb + 3) * 16 + ln) * DD;
    for (int db = 0; db < DD; db += 32) {
      v16h a  = afrag(qrow + db, hs);
      v16h b0 = bfrag(k0 + db, hs);
      v16h b1 = bfrag(k1 + db, hs);
      v16h b2 = bfrag(k2 + db, hs);
      v16h b3 = bfrag(k3 + db, hs);
      s0 = WMMA_F16(a, b0, s0);
      s1 = WMMA_F16(a, b1, s1);
      s2 = WMMA_F16(a, b2, s2);
      s3 = WMMA_F16(a, b3, s3);
    }
    for (int r = 0; r < 8; ++r) {
      const int q = (r + hs * 8) * TT;
      S[q + (ktb + 0) * 16 + ln] = s0[r] * scale;
      S[q + (ktb + 1) * 16 + ln] = s1[r] * scale;
      S[q + (ktb + 2) * 16 + ln] = s2[r] * scale;
      S[q + (ktb + 3) * 16 + ln] = s3[r] * scale;
    }
  }
  __syncthreads();

  // ---- Phase 2: masked softmax over keys (16 rows x 16 threads each) ----
  const unsigned char* mb = mask + (size_t)b * TT;
  const int row = tid >> 4, sub = tid & 15;
  float lm = NEG_INF;
  for (int k = sub; k < TT; k += 16)
    if (mb[k]) lm = fmaxf(lm, S[row * TT + k]);
  red[row * 16 + sub] = lm;
  __syncthreads();
  float rm = NEG_INF;
  for (int i = 0; i < 16; ++i) rm = fmaxf(rm, red[row * 16 + i]);
  __syncthreads();
  float ls = 0.0f;
  for (int k = sub; k < TT; k += 16) {
    float e = mb[k] ? __expf(S[row * TT + k] - rm) : 0.0f;
    Ph[row * TT + k] = (_Float16)e;
    ls += e;
  }
  red[row * 16 + sub] = ls;
  __syncthreads();
  if (tid < 16) {
    float t = 0.0f;
    for (int i = 0; i < 16; ++i) t += red[tid * 16 + i];
    rowsum[tid] = t;
  }
  __syncthreads();

  // ---- Phase 3: O = P . V (wave owns 64 d-columns), normalize, mean over q ----
  const int n0 = wave * 64;
  v8f a0 = {}, a1 = {}, a2 = {}, a3 = {};
  const _Float16* vb = Vt + (size_t)b * DD * TT;
  const _Float16* v0 = vb + (size_t)(n0 +  0 + ln) * TT;
  const _Float16* v1 = vb + (size_t)(n0 + 16 + ln) * TT;
  const _Float16* v2 = vb + (size_t)(n0 + 32 + ln) * TT;
  const _Float16* v3 = vb + (size_t)(n0 + 48 + ln) * TT;
  for (int kk = 0; kk < TT; kk += 32) {
    V16U u;
    u.h[0] = *(const v8h*)&Ph[ln * TT + kk + hs * 8];
    u.h[1] = *(const v8h*)&Ph[ln * TT + kk + 16 + hs * 8];
    v16h a  = u.v;
    v16h b0 = bfrag(v0 + kk, hs);
    v16h b1 = bfrag(v1 + kk, hs);
    v16h b2 = bfrag(v2 + kk, hs);
    v16h b3 = bfrag(v3 + kk, hs);
    a0 = WMMA_F16(a, b0, a0);
    a1 = WMMA_F16(a, b1, a1);
    a2 = WMMA_F16(a, b2, a2);
    a3 = WMMA_F16(a, b3, a3);
  }

  float inv[8];
  for (int r = 0; r < 8; ++r) inv[r] = 1.0f / rowsum[r + hs * 8];
  float c0 = 0.f, c1 = 0.f, c2 = 0.f, c3 = 0.f;
  for (int r = 0; r < 8; ++r) {
    c0 += a0[r] * inv[r];
    c1 += a1[r] * inv[r];
    c2 += a2[r] * inv[r];
    c3 += a3[r] * inv[r];
  }
  // add the other 8 query rows held by the opposite 16-lane half (same column)
  c0 += __shfl_xor(c0, 16, 32);
  c1 += __shfl_xor(c1, 16, 32);
  c2 += __shfl_xor(c2, 16, 32);
  c3 += __shfl_xor(c3, 16, 32);
  if (lane < 16) {
    const float invT = 1.0f / (float)TT;
    atomicAdd(&out[(size_t)b * DD + n0 +  0 + ln], c0 * invT);
    atomicAdd(&out[(size_t)b * DD + n0 + 16 + ln], c1 * invT);
    atomicAdd(&out[(size_t)b * DD + n0 + 32 + ln], c2 * invT);
    atomicAdd(&out[(size_t)b * DD + n0 + 48 + ln], c3 * invT);
  }
}

// ---------------- launcher ----------------
extern "C" void kernel_launch(void* const* d_in, const int* in_sizes, int n_in,
                              void* d_out, int out_size, void* d_ws, size_t ws_size,
                              hipStream_t stream) {
  const float*         x     = (const float*)d_in[0];
  const unsigned char* mask  = (const unsigned char*)d_in[1];
  const float*         Wq    = (const float*)d_in[2];
  const float*         bq    = (const float*)d_in[3];
  const float*         Wk    = (const float*)d_in[4];
  const float*         bk    = (const float*)d_in[5];
  const float*         Wv    = (const float*)d_in[6];
  const float*         bv    = (const float*)d_in[7];
  const float*         scale = (const float*)d_in[8];
  float*               out   = (float*)d_out;

  char* ws = (char*)d_ws;
  _Float16* xh = (_Float16*)(ws);                                  // 64 MiB
  _Float16* Qh = (_Float16*)(ws + (size_t)64  * 1024 * 1024);      // 64 MiB
  _Float16* Kh = (_Float16*)(ws + (size_t)128 * 1024 * 1024);      // 64 MiB
  _Float16* Vt = (_Float16*)(ws + (size_t)192 * 1024 * 1024);      // 64 MiB (transposed [b][d][t])
  _Float16* Wt = (_Float16*)(ws + (size_t)256 * 1024 * 1024);      // 1.5 MiB

  const int nx = MTOT * DD; // 33,554,432
  k_zero_out<<<(BB * DD + 255) / 256, 256, 0, stream>>>(out, BB * DD);
  k_cvt_x<<<(nx / 4 + 255) / 256, 256, 0, stream>>>(x, xh, nx);
  k_cvt_w<<<(3 * DD * DD + 255) / 256, 256, 0, stream>>>(Wq, Wk, Wv, Wt);

  dim3 gqkv(MTOT / 256, DD / 64, 3);
  k_qkv<<<gqkv, 256, 0, stream>>>(xh, Wt, bq, bk, bv, Qh, Kh, Vt);

  k_attn<<<BB * (TT / 16), 256, 0, stream>>>(Qh, Kh, Vt, mask, scale, out);
}
